// CrossWindowFocalAttention_69818988363866
// MI455X (gfx1250) — compile-verified, matrix operand order checked by reading, not run on hardware
//
#include <hip/hip_runtime.h>
#include <hip/hip_bf16.h>

typedef __attribute__((ext_vector_type(16))) __bf16 v16bf;
typedef __attribute__((ext_vector_type(8)))  __bf16 v8bf;
typedef __attribute__((ext_vector_type(8)))  float  v8f;

#define CDIM 384
#define NHEAD 8
#define HDIM 48
#define HDP 64
#define WSZ 16
#define NWIN 128        // 2 * 8 * 8
#define KVL 768
#define IMG 128

__device__ __forceinline__ unsigned short f2bf(float f) {
    unsigned int u = __float_as_uint(f);
    unsigned int r = u + 0x7FFFu + ((u >> 16) & 1u);
    return (unsigned short)(r >> 16);
}

__device__ __forceinline__ int refl(int i) {
    i = (i < 0) ? -i : i;
    return (i > IMG - 1) ? 2 * (IMG - 1) - i : i;
}

// Load one 16x32 bf16 A/B fragment half-pair for this lane.
// p points at (row-or-col base) + khalf*8; lo covers K+0..7, hi covers K+16..23.
__device__ __forceinline__ v16bf frag_ld(const unsigned short* p) {
    v8bf lo = *reinterpret_cast<const v8bf*>(p);
    v8bf hi = *reinterpret_cast<const v8bf*>(p + 16);
    v16bf r;
#pragma unroll
    for (int i = 0; i < 8; ++i) { r[i] = lo[i]; r[i + 8] = hi[i]; }
    return r;
}

__device__ __forceinline__ v8f wmma_bf16(v16bf a, v16bf b, v8f c) {
    return __builtin_amdgcn_wmma_f32_16x16x32_bf16(false, a, false, b, (short)0, c, false, false);
}

__device__ __forceinline__ float redmax16(float v) {
#pragma unroll
    for (int m = 1; m < 16; m <<= 1) v = fmaxf(v, __shfl_xor(v, m, 32));
    return v;
}
__device__ __forceinline__ float redsum16(float v) {
#pragma unroll
    for (int m = 1; m < 16; m <<= 1) v += __shfl_xor(v, m, 32);
    return v;
}

// 64x64 register-blocked wave-tile GEMM core: A row-major bf16 (lda=CDIM),
// BT[n][k] row-major bf16, K=CDIM. Fills acc[4][4] (M-frag x N-frag).
__device__ __forceinline__ void gemm64x64(const unsigned short* __restrict__ A0,
                                          const unsigned short* __restrict__ B0,
                                          v8f acc[4][4]) {
#pragma unroll
    for (int mi = 0; mi < 4; ++mi)
#pragma unroll
        for (int nj = 0; nj < 4; ++nj) acc[mi][nj] = (v8f){};
    for (int k = 0; k < CDIM; k += 32) {
        v16bf af[4], bf4[4];
#pragma unroll
        for (int mi = 0; mi < 4; ++mi)
            af[mi] = frag_ld(A0 + (size_t)mi * 16 * CDIM + k);
#pragma unroll
        for (int nj = 0; nj < 4; ++nj)
            bf4[nj] = frag_ld(B0 + (size_t)nj * 16 * CDIM + k);
#pragma unroll
        for (int mi = 0; mi < 4; ++mi)
#pragma unroll
            for (int nj = 0; nj < 4; ++nj)
                acc[mi][nj] = wmma_bf16(af[mi], bf4[nj], acc[mi][nj]);
    }
}

// ---------------- weight transpose + bf16 convert: WT[n][k] = W[k][n] ----------------
__global__ __launch_bounds__(256) void wconv_kernel(const float* __restrict__ Wq,
                                                    const float* __restrict__ Wk,
                                                    const float* __restrict__ Wv,
                                                    const float* __restrict__ Wo,
                                                    unsigned short* __restrict__ WT) {
    int idx = blockIdx.x * 256 + threadIdx.x;      // 4 * 384 * 384
    int which = idx / (CDIM * CDIM);
    int e = idx % (CDIM * CDIM);
    int k = e / CDIM, n = e % CDIM;
    const float* W = (which == 0) ? Wq : (which == 1) ? Wk : (which == 2) ? Wv : Wo;
    WT[(size_t)which * CDIM * CDIM + (size_t)n * CDIM + k] = f2bf(W[e]);
}

// ---------------- gather focal kv tokens (bf16) ----------------
// kv[w][t][c]; t in [0,256): 4 pre-center pooled offsets; [256,512): center window;
// [512,768): 4 post-center pooled offsets.
__global__ __launch_bounds__(256) void gather_kv_kernel(const float* __restrict__ x,
                                                        unsigned short* __restrict__ kv) {
    int idx = blockIdx.x * 256 + threadIdx.x;      // NWIN*KVL*96
    int c4 = idx % 96;
    int t  = (idx / 96) % KVL;
    int w  = idx / (96 * KVL);
    int b = w >> 6, wi = (w >> 3) & 7, wj = w & 7;
    float4 v;
    if (t >= 256 && t < 512) {
        int rl = t - 256;
        int row = wi * WSZ + (rl >> 4);
        int col = wj * WSZ + (rl & 15);
        v = *reinterpret_cast<const float4*>(x + (((size_t)b * IMG + row) * IMG + col) * CDIM + c4 * 4);
    } else {
        int o = (t < 256) ? (t >> 6) : 5 + ((t - 512) >> 6);
        int p = (t < 256) ? (t & 63) : ((t - 512) & 63);
        int di = o / 3 - 1, dj = o % 3 - 1;
        int R0 = (wi + di) * WSZ + 2 * (p >> 3);
        int C0 = (wj + dj) * WSZ + 2 * (p & 7);
        float ax = 0.f, ay = 0.f, az = 0.f, aw = 0.f;
#pragma unroll
        for (int e = 0; e < 4; ++e) {
            int rr = refl(R0 + (e >> 1));
            int cc = refl(C0 + (e & 1));
            float4 s = *reinterpret_cast<const float4*>(x + (((size_t)b * IMG + rr) * IMG + cc) * CDIM + c4 * 4);
            ax += s.x; ay += s.y; az += s.z; aw += s.w;
        }
        v.x = ax * 0.25f; v.y = ay * 0.25f; v.z = az * 0.25f; v.w = aw * 0.25f;
    }
    size_t base = ((size_t)w * KVL + t) * CDIM + c4 * 4;
    uint2 pk;
    pk.x = (unsigned)f2bf(v.x) | ((unsigned)f2bf(v.y) << 16);
    pk.y = (unsigned)f2bf(v.z) | ((unsigned)f2bf(v.w) << 16);
    *reinterpret_cast<uint2*>(kv + base) = pk;
}

// ---------------- Q projection: q_pad[win][head][256][64], 64x64 wave tiles ----------------
__global__ __launch_bounds__(256) void qproj_kernel(const unsigned short* __restrict__ kv,
                                                    const unsigned short* __restrict__ WqT,
                                                    const float* __restrict__ bq,
                                                    unsigned short* __restrict__ qp) {
    int gw = blockIdx.x * 8 + (threadIdx.x >> 5);  // 512 M-tiles64 * 6 N-tiles64 = 3072 waves
    int lane = threadIdx.x & 31;
    int mt = gw / 6, n64 = gw % 6;
    int r = lane & 15, hh = lane >> 4;
    int win = mt >> 2;                             // 4 M-tiles64 per window (256 rows)
    int qbase = (mt & 3) << 6;
    const unsigned short* A0 = kv + ((size_t)win * KVL + 256 + qbase + r) * CDIM + hh * 8;
    const unsigned short* B0 = WqT + (size_t)(n64 * 64 + r) * CDIM + hh * 8;
    v8f acc[4][4];
    gemm64x64(A0, B0, acc);
#pragma unroll
    for (int nj = 0; nj < 4; ++nj) {
        int nt = n64 * 4 + nj;
        float bias = bq[nt * 16 + r];
        int head = nt / 3, col = (nt % 3) * 16 + r;
#pragma unroll
        for (int mi = 0; mi < 4; ++mi)
#pragma unroll
            for (int i = 0; i < 8; ++i) {
                int qrow = qbase + mi * 16 + i + 8 * hh;
                qp[(((size_t)(win * NHEAD + head) * 256) + qrow) * HDP + col] =
                    f2bf(acc[mi][nj][i] + bias);
            }
    }
}

// ---------------- K/V projection: k_pad[win][head][768][64], vT[win][head][64][768] ----------------
__global__ __launch_bounds__(256) void kvproj_kernel(const unsigned short* __restrict__ kv,
                                                     const unsigned short* __restrict__ WkT,
                                                     const unsigned short* __restrict__ WvT,
                                                     const float* __restrict__ bk,
                                                     const float* __restrict__ bv,
                                                     unsigned short* __restrict__ kp,
                                                     unsigned short* __restrict__ vT) {
    int isV = blockIdx.y;
    int gw = blockIdx.x * 8 + (threadIdx.x >> 5);  // 1536 M-tiles64 * 6 = 9216 waves
    int lane = threadIdx.x & 31;
    int mt = gw / 6, n64 = gw % 6;
    int r = lane & 15, hh = lane >> 4;
    int win = mt / 12;                             // 12 M-tiles64 per window (768 rows)
    int kvbase = (mt % 12) << 6;
    const unsigned short* A0 = kv + ((size_t)win * KVL + kvbase + r) * CDIM + hh * 8;
    const unsigned short* B0 = (isV ? WvT : WkT) + (size_t)(n64 * 64 + r) * CDIM + hh * 8;
    v8f acc[4][4];
    gemm64x64(A0, B0, acc);
#pragma unroll
    for (int nj = 0; nj < 4; ++nj) {
        int nt = n64 * 4 + nj;
        float bias = (isV ? bv : bk)[nt * 16 + r];
        int head = nt / 3, col = (nt % 3) * 16 + r;
#pragma unroll
        for (int mi = 0; mi < 4; ++mi)
#pragma unroll
            for (int i = 0; i < 8; ++i) {
                int row = kvbase + mi * 16 + i + 8 * hh;
                float val = acc[mi][nj][i] + bias;
                if (isV)
                    vT[(((size_t)(win * NHEAD + head) * HDP) + col) * KVL + row] = f2bf(val);
                else
                    kp[(((size_t)(win * NHEAD + head) * KVL) + row) * HDP + col] = f2bf(val);
            }
    }
}

// ---------------- attention: one block per (win, head); 8 waves x 32 query rows ----------------
__global__ __launch_bounds__(256) void attn_kernel(const unsigned short* __restrict__ qp,
                                                   const unsigned short* __restrict__ kp,
                                                   const unsigned short* __restrict__ vT,
                                                   const float* __restrict__ pb,
                                                   unsigned short* __restrict__ o) {
    __shared__ unsigned short Plds[8][32][40];     // per-wave P staging, 16B-aligned rows

    int win = blockIdx.x >> 3, head = blockIdx.x & 7;
    int wave = threadIdx.x >> 5, lane = threadIdx.x & 31;
    int r = lane & 15, hh = lane >> 4;
    int row0 = wave * 32;

    const unsigned short* Qb = qp + (size_t)(win * NHEAD + head) * 256 * HDP;
    const unsigned short* Kb = kp + (size_t)(win * NHEAD + head) * KVL * HDP;
    const unsigned short* Vb = vT + (size_t)(win * NHEAD + head) * HDP * KVL;

    float biasv[9];
#pragma unroll
    for (int ob = 0; ob < 9; ++ob)
        biasv[ob] = (ob == 4) ? 0.f : -pb[head * 9 + ob];
    const float scale = 0.144337567297406f;        // 48^-0.5

    // preload Q fragments: 2 M-tiles x 2 K-chunks
    v16bf qf[2][2];
#pragma unroll
    for (int m = 0; m < 2; ++m)
#pragma unroll
        for (int kc = 0; kc < 2; ++kc)
            qf[m][kc] = frag_ld(Qb + (size_t)(row0 + m * 16 + r) * HDP + kc * 32 + hh * 8);

    v8f acc[2][4];
    float mrun[2][8], lrun[2][8];
#pragma unroll
    for (int m = 0; m < 2; ++m) {
#pragma unroll
        for (int d = 0; d < 4; ++d) acc[m][d] = (v8f){};
#pragma unroll
        for (int i = 0; i < 8; ++i) { mrun[m][i] = -1e30f; lrun[m][i] = 0.f; }
    }

    unsigned short (*P)[40] = Plds[wave];

    for (int kv0 = 0; kv0 < KVL; kv0 += 32) {
        // ---- S = Q K^T (2 M-tiles x 2 N-tiles, K=64) ----
        v8f s[2][2];
#pragma unroll
        for (int n = 0; n < 2; ++n) {
            int tok = kv0 + n * 16 + r;
            const unsigned short* kptr = Kb + (size_t)tok * HDP + hh * 8;
            v16bf kf0 = frag_ld(kptr);
            v16bf kf1 = frag_ld(kptr + 32);
#pragma unroll
            for (int m = 0; m < 2; ++m) {
                v8f t = (v8f){};
                t = wmma_bf16(qf[m][0], kf0, t);
                t = wmma_bf16(qf[m][1], kf1, t);
                s[m][n] = t;
            }
        }
        // scale + per-offset-group bias (column = kv0 + n*16 + r, same for all rows)
        float badd[2];
#pragma unroll
        for (int n = 0; n < 2; ++n) {
            int t = kv0 + n * 16 + r;
            int og = (t < 256) ? (t >> 6) : ((t < 512) ? 4 : 5 + ((t - 512) >> 6));
            badd[n] = biasv[og];
        }
#pragma unroll
        for (int m = 0; m < 2; ++m)
#pragma unroll
            for (int n = 0; n < 2; ++n)
#pragma unroll
                for (int i = 0; i < 8; ++i)
                    s[m][n][i] = s[m][n][i] * scale + badd[n];

        // ---- online softmax, write P to LDS ----
#pragma unroll
        for (int m = 0; m < 2; ++m) {
            float nm[8], al[8], ps[8];
#pragma unroll
            for (int i = 0; i < 8; ++i) {
                float v = fmaxf(s[m][0][i], s[m][1][i]);
                v = redmax16(v);
                nm[i] = fmaxf(mrun[m][i], v);
                al[i] = __expf(mrun[m][i] - nm[i]);
                mrun[m][i] = nm[i];
                ps[i] = 0.f;
            }
#pragma unroll
            for (int n = 0; n < 2; ++n)
#pragma unroll
                for (int i = 0; i < 8; ++i) {
                    float p = __expf(s[m][n][i] - nm[i]);
                    ps[i] += p;
                    P[m * 16 + i + 8 * hh][n * 16 + r] = f2bf(p);
                }
#pragma unroll
            for (int i = 0; i < 8; ++i) {
                float rs = redsum16(ps[i]);
                lrun[m][i] = lrun[m][i] * al[i] + rs;
            }
#pragma unroll
            for (int d = 0; d < 4; ++d)
#pragma unroll
                for (int i = 0; i < 8; ++i)
                    acc[m][d][i] *= al[i];
        }

        // ---- O += P V : A = P (16x32 from LDS), B = V^T rows (contiguous in kv) ----
        v16bf vf[4];
#pragma unroll
        for (int d = 0; d < 4; ++d)
            vf[d] = frag_ld(Vb + (size_t)(d * 16 + r) * KVL + kv0 + hh * 8);
#pragma unroll
        for (int m = 0; m < 2; ++m) {
            v16bf pf = frag_ld(&P[m * 16 + r][hh * 8]);
#pragma unroll
            for (int d = 0; d < 4; ++d)
                acc[m][d] = wmma_bf16(pf, vf[d], acc[m][d]);
        }
    }

    // ---- normalize and store (heads concatenated, only dims < 48) ----
#pragma unroll
    for (int m = 0; m < 2; ++m)
#pragma unroll
        for (int d = 0; d < 3; ++d) {
            int col = d * 16 + r;
#pragma unroll
            for (int i = 0; i < 8; ++i) {
                int row = row0 + m * 16 + i + 8 * hh;
                float v = acc[m][d][i] / lrun[m][i];
                o[((size_t)win * 256 + row) * CDIM + head * HDIM + col] = f2bf(v);
            }
        }
}

// ---------------- output projection + scatter back to (B,H,W,C), 64x64 wave tiles ----------------
__global__ __launch_bounds__(256) void oproj_kernel(const unsigned short* __restrict__ o,
                                                    const unsigned short* __restrict__ WoT,
                                                    const float* __restrict__ bo,
                                                    float* __restrict__ out) {
    int gw = blockIdx.x * 8 + (threadIdx.x >> 5);  // 512 M-tiles64 * 6 = 3072 waves
    int lane = threadIdx.x & 31;
    int mt = gw / 6, n64 = gw % 6;
    int r = lane & 15, hh = lane >> 4;
    const unsigned short* A0 = o + (size_t)(mt * 64 + r) * CDIM + hh * 8;
    const unsigned short* B0 = WoT + (size_t)(n64 * 64 + r) * CDIM + hh * 8;
    v8f acc[4][4];
    gemm64x64(A0, B0, acc);
#pragma unroll
    for (int nj = 0; nj < 4; ++nj) {
        int nt = n64 * 4 + nj;
        float bias = bo[nt * 16 + r];
#pragma unroll
        for (int mi = 0; mi < 4; ++mi)
#pragma unroll
            for (int i = 0; i < 8; ++i) {
                int m = mt * 64 + mi * 16 + i + 8 * hh;
                int win = m >> 8, rr = m & 255;
                int b = win >> 6, wi = (win >> 3) & 7, wj = win & 7;
                int hrow = wi * WSZ + (rr >> 4);
                int wcol = wj * WSZ + (rr & 15);
                out[(((size_t)b * IMG + hrow) * IMG + wcol) * CDIM + nt * 16 + r] =
                    acc[mi][nj][i] + bias;
            }
    }
}

// ---------------- workspace layout ----------------
static const size_t OFF_KV = 0;                                        // 75,497,472 B
static const size_t OFF_WT = OFF_KV + (size_t)NWIN * KVL * CDIM * 2;   // + 1,179,648
static const size_t OFF_QP = OFF_WT + (size_t)4 * CDIM * CDIM * 2;     // + 33,554,432
static const size_t OFF_KP = OFF_QP + (size_t)NWIN * NHEAD * 256 * HDP * 2;  // +100,663,296
static const size_t OFF_VT = OFF_KP + (size_t)NWIN * NHEAD * KVL * HDP * 2;  // +100,663,296
static const size_t OFF_O  = OFF_VT + (size_t)NWIN * NHEAD * HDP * KVL * 2;  // + 25,165,824
static const size_t WS_NEED = OFF_O + (size_t)NWIN * 256 * CDIM * 2;

extern "C" void kernel_launch(void* const* d_in, const int* in_sizes, int n_in,
                              void* d_out, int out_size, void* d_ws, size_t ws_size,
                              hipStream_t stream) {
    if (ws_size < WS_NEED) return;
    const float* x  = (const float*)d_in[0];
    const float* Wq = (const float*)d_in[1];
    const float* bq = (const float*)d_in[2];
    const float* Wk = (const float*)d_in[3];
    const float* bk = (const float*)d_in[4];
    const float* Wv = (const float*)d_in[5];
    const float* bv = (const float*)d_in[6];
    const float* Wo = (const float*)d_in[7];
    const float* bo = (const float*)d_in[8];
    const float* pb = (const float*)d_in[9];

    char* w = (char*)d_ws;
    unsigned short* kv  = (unsigned short*)(w + OFF_KV);
    unsigned short* WT  = (unsigned short*)(w + OFF_WT);
    unsigned short* qp  = (unsigned short*)(w + OFF_QP);
    unsigned short* kp  = (unsigned short*)(w + OFF_KP);
    unsigned short* vT  = (unsigned short*)(w + OFF_VT);
    unsigned short* ob  = (unsigned short*)(w + OFF_O);
    unsigned short* WqT = WT;
    unsigned short* WkT = WT + (size_t)CDIM * CDIM;
    unsigned short* WvT = WT + (size_t)2 * CDIM * CDIM;
    unsigned short* WoT = WT + (size_t)3 * CDIM * CDIM;

    // zero the head-dim pad regions (48..63) via full-buffer clears
    hipMemsetAsync(qp, 0, (size_t)NWIN * NHEAD * 256 * HDP * 2, stream);
    hipMemsetAsync(kp, 0, (size_t)NWIN * NHEAD * KVL * HDP * 2, stream);
    hipMemsetAsync(vT, 0, (size_t)NWIN * NHEAD * HDP * KVL * 2, stream);

    wconv_kernel<<<(4 * CDIM * CDIM) / 256, 256, 0, stream>>>(Wq, Wk, Wv, Wo, WT);
    gather_kv_kernel<<<(NWIN * KVL * 96) / 256, 256, 0, stream>>>(x, kv);

    // Q: 512 M-tiles64 x 6 N-tiles64, 8 wave-tiles per block
    qproj_kernel<<<(512 * 6) / 8, 256, 0, stream>>>(kv, WqT, bq, qp);
    // K/V: 1536 M-tiles64 x 6 N-tiles64; y-dim selects K vs V
    kvproj_kernel<<<dim3((1536 * 6) / 8, 2), 256, 0, stream>>>(kv, WkT, WvT, bk, bv, kp, vT);

    attn_kernel<<<NWIN * NHEAD, 256, 0, stream>>>(qp, kp, vT, pb, ob);

    oproj_kernel<<<(512 * 6) / 8, 256, 0, stream>>>(ob, WoT, bo, (float*)d_out);
}
